// self_attn_62715112457016
// MI455X (gfx1250) — matrix operand order repaired and checked by reading, not compiled, not run on hardware
//
#include <hip/hip_runtime.h>
#include <hip/hip_bf16.h>

typedef _Float16 h16;
typedef __attribute__((ext_vector_type(8)))  _Float16 v8h;
typedef __attribute__((ext_vector_type(16))) _Float16 v16h;
typedef __attribute__((ext_vector_type(8)))  float    v8f;

// ---------------------------------------------------------------------------
// D = A(16x32 f16) * B(32x16 f16) + C(16x16 f32)   (wave32 WMMA)
// ---------------------------------------------------------------------------
__device__ __forceinline__ v8f wmma16(v16h a, v16h b, v8f c) {
  return __builtin_amdgcn_wmma_f32_16x16x32_f16(
      /*neg_a=*/false, a, /*neg_b=*/false, b,
      /*c_mod=*/(short)0, c, /*reuse_a=*/false, /*reuse_b=*/false);
}

// fast sigmoid: v_exp_f32 + v_add + v_rcp_f32 (no IEEE division sequence)
__device__ __forceinline__ float fast_sigmoid(float s) {
  return __builtin_amdgcn_rcpf(1.0f + __expf(-s));
}

// A-matrix fragment (16x32, M x K) from row-major [row][k] storage.
// ISA layout: lanes 0-15 row=lane hold K={0..7,16..23}; lanes 16-31 hold
// K={8..15,24..31}.  Two 16B contiguous groups per lane.
__device__ __forceinline__ v16h load_frag_a(const h16* base, int ld,
                                            int row0, int k0, int lane) {
  int r  = row0 + (lane & 15);
  int kb = (lane & 16) ? 8 : 0;
  const h16* p = base + (size_t)r * (size_t)ld + k0 + kb;
  v8h lo = *(const v8h*)(p);
  v8h hi = *(const v8h*)(p + 16);
  v16h a;
#pragma unroll
  for (int e = 0; e < 8; ++e) { a[e] = lo[e]; a[e + 8] = hi[e]; }
  return a;
}

// B-matrix fragment (32x16, K x N) gathered from B-transposed row-major
// [n][k] storage.  ISA layout: lanes 0-15 hold K={0..7,16..23} of column n=lane,
// lanes 16-31 hold K={8..15,24..31} of column n=lane-16.  Two 16B groups per lane.
__device__ __forceinline__ v16h load_frag_b(const h16* baseT, int ld,
                                            int col0, int k0, int lane) {
  int c  = col0 + (lane & 15);
  int kb = (lane & 16) ? 8 : 0;
  const h16* p = baseT + (size_t)c * (size_t)ld + k0 + kb;
  v8h lo = *(const v8h*)(p);
  v8h hi = *(const v8h*)(p + 16);
  v16h b;
#pragma unroll
  for (int e = 0; e < 8; ++e) { b[e] = lo[e]; b[e + 8] = hi[e]; }
  return b;
}

// ---------------------------------------------------------------------------
// Kernel 1: out = W[M x 512] * x[b][512 x 4096], converted to f16.
// TRANSP=true  -> out[b][n][M]   (for q^T / k^T, M = 64)
// TRANSP=false -> out[b][M][n]   (for v,        M = 512)
// Workgroup: 256 threads (8 waves), 64(o) x 64(n) tile, K chunks of 32.
// LDS double-buffered -> one barrier per K-chunk.
// ---------------------------------------------------------------------------
template <bool TRANSP>
__global__ __launch_bounds__(256) void gemm_wx_kernel(
    const float* __restrict__ W,   // [M][512]
    const float* __restrict__ x,   // [B][512][4096]
    h16* __restrict__ out, int M) {
  constexpr int K = 512, N = 4096;
  __shared__ h16 Ws[2][64 * 32];   // W chunk (f16), double-buffered
  __shared__ h16 XsT[2][64 * 48];  // x chunk transposed [n][c], padded stride

  const int b    = blockIdx.z;
  const int o0   = blockIdx.y * 64;
  const int n0   = blockIdx.x * 64;
  const int tid  = threadIdx.x;
  const int lane = tid & 31;
  const int wave = tid >> 5;
  const int ot   = wave & 3;            // this wave's o-tile
  const int nt0  = (wave >> 2) * 2;     // this wave's pair of n-tiles

  const float* xb = x + (size_t)b * K * N;

  v8f acc0 = {}, acc1 = {};

#pragma unroll 1
  for (int c0 = 0; c0 < K; c0 += 32) {
    const int buf = (c0 >> 5) & 1;
    {  // stage W[o0..o0+63][c0..c0+31] -> f16 LDS
      int row = tid >> 2;
      int col = (tid & 3) * 8;
      const float* wp = W + (size_t)(o0 + row) * K + c0 + col;
      float4 f0 = *(const float4*)wp;
      float4 f1 = *(const float4*)(wp + 4);
      v8h hv;
      hv[0] = (h16)f0.x; hv[1] = (h16)f0.y; hv[2] = (h16)f0.z; hv[3] = (h16)f0.w;
      hv[4] = (h16)f1.x; hv[5] = (h16)f1.y; hv[6] = (h16)f1.z; hv[7] = (h16)f1.w;
      *(v8h*)&Ws[buf][row * 32 + col] = hv;
    }
    {  // stage x[c0..c0+31][n0..n0+63] transposed -> XsT[n][c]
      int cl = tid >> 3;
      int nl = (tid & 7) * 8;
      const float* xp = xb + (size_t)(c0 + cl) * N + n0 + nl;
      float4 f0 = *(const float4*)xp;
      float4 f1 = *(const float4*)(xp + 4);
      float f[8] = {f0.x, f0.y, f0.z, f0.w, f1.x, f1.y, f1.z, f1.w};
#pragma unroll
      for (int e = 0; e < 8; ++e) XsT[buf][(nl + e) * 48 + cl] = (h16)f[e];
    }
    __syncthreads();   // single barrier: next iter writes the other buffer

    v16h a  = load_frag_a(Ws[buf], 32, ot * 16, 0, lane);
    v16h b0 = load_frag_b(XsT[buf], 48, nt0 * 16, 0, lane);
    v16h b1 = load_frag_b(XsT[buf], 48, (nt0 + 1) * 16, 0, lane);
    acc0 = wmma16(a, b0, acc0);
    acc1 = wmma16(a, b1, acc1);
  }

  // store per ISA C/D layout: lane<16 -> n=lane, m=v ; lane>=16 -> n=lane-16, m=v+8
  const int nA = n0 + nt0 * 16 + (lane & 15);
  const int nB = nA + 16;
  const int ob = o0 + ot * 16 + ((lane & 16) ? 8 : 0);
  if (TRANSP) {
    v8h h0, h1;
#pragma unroll
    for (int v = 0; v < 8; ++v) { h0[v] = (h16)acc0[v]; h1[v] = (h16)acc1[v]; }
    *(v8h*)(out + ((size_t)b * N + nA) * M + ob) = h0;  // 8 consecutive o's
    *(v8h*)(out + ((size_t)b * N + nB) * M + ob) = h1;
  } else {
#pragma unroll
    for (int v = 0; v < 8; ++v) {
      out[((size_t)b * M + ob + v) * N + nA] = (h16)acc0[v];
      out[((size_t)b * M + ob + v) * N + nB] = (h16)acc1[v];
    }
  }
}

// ---------------------------------------------------------------------------
// Kernel 2: fused sigmoid-attention.
//   S[i,j] = q[:,i].k[:,j]  (WMMA, K=64)    P = sigmoid(S) -> f16 in LDS
//   O[c,i] += v[c,j-block] * P^T            (WMMA, K=64 per j-step)
//   out = gamma * O + x
// WG = (batch b, 64-wide i-block); 8 waves; each wave owns 64(c) x 64(i) of O.
// P is double-buffered in LDS -> exactly one barrier per j-iteration.
// ---------------------------------------------------------------------------
__global__ __launch_bounds__(256) void attn_kernel(
    const h16* __restrict__ qT,   // [B][N][64]
    const h16* __restrict__ kT,   // [B][N][64]
    const h16* __restrict__ vh,   // [B][512][N]
    const float* __restrict__ x,  // [B][512][N]
    const float* __restrict__ gamma_p,
    float* __restrict__ out) {
  constexpr int N = 4096, C = 512, CQ = 64, BI = 64, BJ = 64;
  __shared__ h16 P[2][BI * BJ];   // double-buffered sigmoid(S) tile, 2 x 8KB

  const int b    = blockIdx.y;
  const int i0   = blockIdx.x * BI;
  const int tid  = threadIdx.x;
  const int lane = tid & 31;
  const int wave = tid >> 5;

  const h16* qb = qT + (size_t)b * N * CQ;
  const h16* kb = kT + (size_t)b * N * CQ;
  const h16* vb = vh + (size_t)b * C * N;

  // S-tile assignment: wave -> (it = wave&3, jt in {2*(wave>>2), +1})
  const int it  = wave & 3;
  const int jt0 = (wave >> 2) * 2;

  // q fragments are loop-invariant for this wave's i-tile
  const v16h qa0 = load_frag_a(qb, CQ, i0 + it * 16, 0, lane);
  const v16h qa1 = load_frag_a(qb, CQ, i0 + it * 16, 32, lane);

  v8f acc[4][4];
  {
    v8f z = {};
#pragma unroll
    for (int a = 0; a < 4; ++a)
#pragma unroll
      for (int c = 0; c < 4; ++c) acc[a][c] = z;
  }

#pragma unroll 1
  for (int j0 = 0; j0 < N; j0 += BJ) {
    h16* Pw = P[(j0 >> 6) & 1];
    if (j0 + BJ < N) {  // warm caches for the next k block
      __builtin_prefetch(kb + (size_t)(j0 + BJ) * CQ, 0, 0);
    }
    // ---- phase 1: P = sigmoid(q^T k); batch all 4 loads, then 4 WMMAs ----
    {
      v16h k0a = load_frag_b(kb, CQ, j0 + jt0 * 16, 0, lane);
      v16h k0b = load_frag_b(kb, CQ, j0 + jt0 * 16, 32, lane);
      v16h k1a = load_frag_b(kb, CQ, j0 + (jt0 + 1) * 16, 0, lane);
      v16h k1b = load_frag_b(kb, CQ, j0 + (jt0 + 1) * 16, 32, lane);
      v8f s0 = {}, s1 = {};
      s0 = wmma16(qa0, k0a, s0);
      s1 = wmma16(qa0, k1a, s1);
      s0 = wmma16(qa1, k0b, s0);
      s1 = wmma16(qa1, k1b, s1);
      const int pj = jt0 * 16 + (lane & 15);
      const int mi = it * 16 + ((lane & 16) ? 8 : 0);
#pragma unroll
      for (int v = 0; v < 8; ++v) {
        Pw[(mi + v) * BJ + pj]      = (h16)fast_sigmoid(s0[v]);
        Pw[(mi + v) * BJ + pj + 16] = (h16)fast_sigmoid(s1[v]);
      }
    }
    __syncthreads();

    // ---- phase 2: O[c-block, i-block] += v[c-block, j-block] * P^T ----
#pragma unroll
    for (int ct = 0; ct < 4; ++ct) {
      const int c = wave * 64 + ct * 16;
      v16h va0 = load_frag_a(vb, N, c, j0, lane);
      v16h va1 = load_frag_a(vb, N, c, j0 + 32, lane);
#pragma unroll
      for (int itl = 0; itl < 4; ++itl) {
        v16h p0 = load_frag_b(Pw, BJ, itl * 16, 0, lane);   // B = P^T
        v16h p1 = load_frag_b(Pw, BJ, itl * 16, 32, lane);
        acc[ct][itl] = wmma16(va0, p0, acc[ct][itl]);
        acc[ct][itl] = wmma16(va1, p1, acc[ct][itl]);
      }
    }
    // no trailing barrier: next iteration writes the other P buffer
  }

  // ---- epilogue: out = gamma * O + x ----
  const float g = gamma_p[0];
#pragma unroll
  for (int ct = 0; ct < 4; ++ct) {
    const int cbase = wave * 64 + ct * 16 + ((lane & 16) ? 8 : 0);
#pragma unroll
    for (int itl = 0; itl < 4; ++itl) {
      const int ii = i0 + itl * 16 + (lane & 15);
#pragma unroll
      for (int v = 0; v < 8; ++v) {
        size_t idx = ((size_t)b * C + cbase + v) * N + ii;
        out[idx] = g * acc[ct][itl][v] + x[idx];
      }
    }
  }
}

// ---------------------------------------------------------------------------
extern "C" void kernel_launch(void* const* d_in, const int* in_sizes, int n_in,
                              void* d_out, int out_size, void* d_ws,
                              size_t ws_size, hipStream_t stream) {
  (void)in_sizes; (void)n_in; (void)out_size; (void)ws_size;
  constexpr int B = 4, C = 512, N = 4096, CQ = 64;

  const float* x     = (const float*)d_in[0];
  const float* Wq    = (const float*)d_in[1];
  const float* Wk    = (const float*)d_in[2];
  const float* Wv    = (const float*)d_in[3];
  const float* gamma = (const float*)d_in[4];
  float* out         = (float*)d_out;

  // workspace: qT[B][N][64], kT[B][N][64], vh[B][512][N]   (all f16, ~21 MB)
  h16* qT = (h16*)d_ws;
  h16* kT = qT + (size_t)B * N * CQ;
  h16* vh = kT + (size_t)B * N * CQ;

  dim3 blk(256);
  gemm_wx_kernel<true ><<<dim3(N / 64, 1,      B), blk, 0, stream>>>(Wq, x, qT, CQ);
  gemm_wx_kernel<true ><<<dim3(N / 64, 1,      B), blk, 0, stream>>>(Wk, x, kT, CQ);
  gemm_wx_kernel<false><<<dim3(N / 64, C / 64, B), blk, 0, stream>>>(Wv, x, vh, C);
  attn_kernel<<<dim3(N / 64, B), blk, 0, stream>>>(qT, kT, vh, x, gamma, out);
}
